// GraphTransformerLayer_12833362280631
// MI455X (gfx1250) — matrix operand order repaired and checked by reading, not compiled, not run on hardware
//
#include <hip/hip_runtime.h>
#include <math.h>
#include <stdint.h>

// ---------------- CDNA5 WMMA types / helpers ----------------
typedef __attribute__((ext_vector_type(16))) _Float16 v16h;
typedef __attribute__((ext_vector_type(8)))  float    v8f;

__device__ __forceinline__ v8f wmma32f16(v16h a, v16h b, v8f c) {
  // v_wmma_f32_16x16x32_f16
  return __builtin_amdgcn_wmma_f32_16x16x32_f16(false, a, false, b, (short)0, c, false, false);
}

// A-matrix (16x32 f16) element (row, kk) -> linear index in a 512-half LDS block,
// laid out so that lane L reads its 16 contiguous halves at [L*16 .. L*16+15].
// Contiguous layout (matches loadB): lane<16 holds row=lane, K in {0..15}; lane>=16 holds
// row=lane-16, K in {16..31}; within lane, half i maps to K = 16*hi + i.
__device__ __forceinline__ int aidx(int row, int kk) {
  int hi = (kk >> 4) & 1;                 // which lane half
  int i  = kk & 15;                       // half index within lane
  return ((row + (hi << 4)) << 4) + i;
}

// B-matrix load: B[k][n] = W[n][k] (row-major W[dout][din], f16).
// Lane<16 holds column n = n0+lane, K = kb*32 + 0..15 ; lane>=16 same n, K += 16.
__device__ __forceinline__ v16h loadB(const _Float16* __restrict__ W, int din,
                                      int n0, int kb, int lane) {
  int n = n0 + (lane & 15);
  int k = (kb << 5) + ((lane >> 4) << 4);
  return *(const v16h*)(W + n * din + k);
}

// CDNA5 async copy: global -> LDS, 16B per lane, tracked by ASYNCcnt.
// Generic LDS pointers carry the LDS byte offset in their low 32 bits
// (flat->LDS aperture mapping truncates to addr[31:0]).
__device__ __forceinline__ void async_tile_load16(const float* gsrc, float* ldst) {
  uint32_t loff = (uint32_t)(uintptr_t)ldst;
  asm volatile("global_load_async_to_lds_b128 %0, %1, off"
               :: "v"(loff), "v"(gsrc)
               : "memory");
}
__device__ __forceinline__ void wait_async0() {
  asm volatile("s_wait_asynccnt 0x0" ::: "memory");
}

// ---------------- weight conversion ----------------
__global__ void __launch_bounds__(256) k_f32_to_f16(const float* __restrict__ src,
                                                    _Float16* __restrict__ dst, int n) {
  int i = blockIdx.x * 256 + threadIdx.x;
  if (i < n) dst[i] = (_Float16)src[i];
}

// ---------------- Q/K/V GEMMs ----------------
// grid (48, 3): 16-row tile of x, which in {Q,K,V}. 8 waves = 8 column tiles of 16.
__global__ void __launch_bounds__(256) k_qkv(
    const _Float16* __restrict__ xh,
    const _Float16* __restrict__ Wq, const _Float16* __restrict__ Wk, const _Float16* __restrict__ Wv,
    const float* __restrict__ bq, const float* __restrict__ bk, const float* __restrict__ bv,
    float* __restrict__ Qs, float* __restrict__ Kf, float* __restrict__ Vf) {
  __shared__ __align__(32) _Float16 aA[2048];   // 4 k-blocks of A layout
  const int m0 = blockIdx.x * 16;
  const int which = blockIdx.y;
  const _Float16* W = (which == 0) ? Wq : (which == 1) ? Wk : Wv;
  const float* bias = (which == 0) ? bq : (which == 1) ? bk : bv;
  float* out = (which == 0) ? Qs : (which == 1) ? Kf : Vf;
  const float scale = (which == 0) ? 0.25f : 1.0f;  // fold 1/sqrt(DF) into Q
  const int tid = threadIdx.x;
  for (int idx = tid; idx < 2048; idx += 256) {
    int row = idx >> 7, k = idx & 127;
    aA[((k >> 5) << 9) + aidx(row, k & 31)] = xh[(m0 + row) * 128 + k];
  }
  __syncthreads();
  const int lane = tid & 31, wid = tid >> 5;
  const int n0 = wid * 16, col = lane & 15, rbase = (lane >> 4) << 3;
  v8f d = {};
#pragma unroll
  for (int kb = 0; kb < 4; ++kb) {
    v16h a = *(const v16h*)(aA + (kb << 9) + lane * 16);
    d = wmma32f16(a, loadB(W, 128, n0, kb, lane), d);
  }
  const int ch = n0 + col;
  const float bb = bias[ch];
#pragma unroll
  for (int r = 0; r < 8; ++r)
    out[(m0 + rbase + r) * 128 + ch] = (d[r] + bb) * scale;
}

// ---------------- fused per-row edge kernel ----------------
// One block per node i. Loop over 48 tiles of 16 j's. Fully fused:
// e -> (E1,E2 WMMA) -> Y -> online softmax + V accum, Y -> e_out WMMA -> LN ->
// FFN-E (2 WMMA stages) -> LN -> E out. 'e' is read once (async, double-buffered
// into LDS via GLOBAL_LOAD_ASYNC_TO_LDS_B128); E written once.
__global__ void __launch_bounds__(256) k_edge(
    const float* __restrict__ e,
    const _Float16* __restrict__ Wemul, const _Float16* __restrict__ Weadd,
    const _Float16* __restrict__ Weout, const _Float16* __restrict__ WlE1,
    const _Float16* __restrict__ WlE2,
    const float* __restrict__ be_mul, const float* __restrict__ be_add,
    const float* __restrict__ be_out, const float* __restrict__ blE1,
    const float* __restrict__ blE2,
    const float* __restrict__ gE1, const float* __restrict__ bE1v,
    const float* __restrict__ gE2, const float* __restrict__ bE2v,
    const float* __restrict__ Qs, const float* __restrict__ Kf, const float* __restrict__ Vf,
    float* __restrict__ wV, float* __restrict__ outE) {
  __shared__ __align__(32) float   Qi[128];
  __shared__ __align__(32) float   sm[128], ss[128], sa[128];  // online softmax state
  __shared__ __align__(32) float   ef32[2][1024]; // double-buffered e tile [16][64] f32
  __shared__ __align__(32) _Float16 eA[1024];    // e tile f16, A layout (2 k-blocks)
  __shared__ __align__(32) _Float16 yA[2048];    // Y tile f16, A layout (4 k-blocks)
  __shared__ __align__(32) float   r64[1024];    // working [16][64] f32
  __shared__ __align__(32) _Float16 e1A[1024];   // e1 f16 A layout
  __shared__ __align__(32) _Float16 hA[1024];    // relu(ff1) f16 A layout

  const int i = blockIdx.x;
  const int tid = threadIdx.x;
  const int lane = tid & 31, wid = tid >> 5;
  const int col = lane & 15, rbase = (lane >> 4) << 3;

  if (tid < 128) { Qi[tid] = Qs[i * 128 + tid]; sm[tid] = -1e30f; ss[tid] = 0.f; sa[tid] = 0.f; }

  // prologue: async-stage tile 0 into buffer 0 (16B per thread, 4KB contiguous)
  async_tile_load16(e + (size_t)(i * 768) * 64 + tid * 4, &ef32[0][tid * 4]);
  __syncthreads();

  // hoisted loop-invariants
  const int c0 = wid * 16;
  const int ch = c0 + col;                 // Y channel (0..127) for this lane
  const float qv = Qi[ch];
  const float bm = be_mul[ch], ba = be_add[ch];
  v16h bmul[2], badd[2];
#pragma unroll
  for (int kb = 0; kb < 2; ++kb) {
    bmul[kb] = loadB(Wemul, 64, c0, kb, lane);
    badd[kb] = loadB(Weadd, 64, c0, kb, lane);
  }
  const int ch64 = c0 + col;               // used by waves 0..3 (0..63)
  float bout = 0.f, bb1 = 0.f, bb2 = 0.f;
  if (wid < 4) { bout = be_out[ch64]; bb1 = blE1[ch64]; bb2 = blE2[ch64]; }
  const int sub = tid & 15, lnedge = tid >> 4, lnc = sub * 4;  // LN: 16 thr/edge
  float g1[4], o1[4], g2[4], o2[4];
#pragma unroll
  for (int c = 0; c < 4; ++c) {
    g1[c] = gE1[lnc + c]; o1[c] = bE1v[lnc + c];
    g2[c] = gE2[lnc + c]; o2[c] = bE2v[lnc + c];
  }

  for (int jt = 0; jt < 48; ++jt) {
    const int j0 = jt * 16;
    const int cur = jt & 1, nxt = cur ^ 1;
    wait_async0();    // tile jt landed in ef32[cur] (own wave's portion)
    __syncthreads();  // all waves' portions landed; prev iteration done with buffers
    if (jt + 1 < 48)  // prefetch tile jt+1 into the other buffer while we compute
      async_tile_load16(e + (size_t)(i * 768 + j0 + 16) * 64 + tid * 4,
                        &ef32[nxt][tid * 4]);
    {  // ---- convert staged e tile -> f16 A layout (LDS -> LDS) ----
      int t4 = tid * 4;
      int row = t4 >> 6, cc = t4 & 63;
      const float4 v = *(const float4*)(&ef32[cur][t4]);
      float vv[4] = {v.x, v.y, v.z, v.w};
#pragma unroll
      for (int c2 = 0; c2 < 4; ++c2) {
        int k = cc + c2;
        eA[((k >> 5) << 9) + aidx(row, k & 31)] = (_Float16)vv[c2];
      }
    }
    __syncthreads();
    {  // ---- E1/E2 WMMA, Y, online softmax + V accumulation, stage Y f16 ----
      v8f d1 = {}, d2 = {};
#pragma unroll
      for (int kb = 0; kb < 2; ++kb) {
        v16h a = *(const v16h*)(eA + (kb << 9) + lane * 16);
        d1 = wmma32f16(a, bmul[kb], d1);
        d2 = wmma32f16(a, badd[kb], d2);
      }
      float y[8];
      float tmax = -1e30f;
#pragma unroll
      for (int r = 0; r < 8; ++r) {
        int jl = rbase + r;
        int jg = j0 + jl;
        float e1v = d1[r] + bm;   // E_mul
        float e2v = d2[r] + ba;   // E_add
        float yv = qv * Kf[jg * 128 + ch] * (e1v + 1.f) + e2v;
        y[r] = yv;
        tmax = fmaxf(tmax, yv);
        yA[((ch >> 5) << 9) + aidx(jl, ch & 31)] = (_Float16)yv;
      }
      tmax = fmaxf(tmax, __shfl_xor(tmax, 16, 32));
      float m_old = sm[ch];
      float m_new = fmaxf(m_old, tmax);
      float tsum = 0.f, tacc = 0.f;
#pragma unroll
      for (int r = 0; r < 8; ++r) {
        int jg = j0 + rbase + r;
        float p = __expf(y[r] - m_new);
        tsum += p;
        tacc += p * Vf[jg * 128 + ch];
      }
      tsum += __shfl_xor(tsum, 16, 32);
      tacc += __shfl_xor(tacc, 16, 32);
      if (lane < 16) {  // one writer per channel
        float sc = __expf(m_old - m_new);
        sm[ch] = m_new;
        ss[ch] = ss[ch] * sc + tsum;
        sa[ch] = sa[ch] * sc + tacc;
      }
    }
    __syncthreads();
    if (wid < 4) {  // ---- newE = Y @ We_out^T + b ; r64 = e + newE ----
      v8f d = {};
#pragma unroll
      for (int kb = 0; kb < 4; ++kb) {
        v16h a = *(const v16h*)(yA + (kb << 9) + lane * 16);
        d = wmma32f16(a, loadB(Weout, 128, c0, kb, lane), d);
      }
#pragma unroll
      for (int r = 0; r < 8; ++r) {
        int idx = (rbase + r) * 64 + ch64;
        r64[idx] = ef32[cur][idx] + d[r] + bout;
      }
    }
    __syncthreads();
    {  // ---- LN1 over 64 channels (16 threads/edge, width-16 shuffles) ----
      float v[4], s1 = 0.f, s2 = 0.f;
#pragma unroll
      for (int c2 = 0; c2 < 4; ++c2) {
        v[c2] = r64[lnedge * 64 + lnc + c2];
        s1 += v[c2]; s2 += v[c2] * v[c2];
      }
#pragma unroll
      for (int o = 1; o < 16; o <<= 1) { s1 += __shfl_xor(s1, o, 16); s2 += __shfl_xor(s2, o, 16); }
      float mu = s1 * (1.f / 64.f);
      float rstd = rsqrtf(s2 * (1.f / 64.f) - mu * mu + 1e-5f);
#pragma unroll
      for (int c2 = 0; c2 < 4; ++c2) {
        float ev = (v[c2] - mu) * rstd * g1[c2] + o1[c2];
        int k = lnc + c2;
        r64[lnedge * 64 + k] = ev;                                  // e1 (f32)
        e1A[((k >> 5) << 9) + aidx(lnedge, k & 31)] = (_Float16)ev; // e1 (f16, A layout)
      }
    }
    __syncthreads();
    if (wid < 4) {  // ---- ff1 = relu(e1 @ WlE1^T + b1) ----
      v8f d = {};
#pragma unroll
      for (int kb = 0; kb < 2; ++kb) {
        v16h a = *(const v16h*)(e1A + (kb << 9) + lane * 16);
        d = wmma32f16(a, loadB(WlE1, 64, c0, kb, lane), d);
      }
#pragma unroll
      for (int r = 0; r < 8; ++r) {
        float hv = fmaxf(d[r] + bb1, 0.f);
        hA[((ch64 >> 5) << 9) + aidx(rbase + r, ch64 & 31)] = (_Float16)hv;
      }
    }
    __syncthreads();
    if (wid < 4) {  // ---- r64 = e1 + ff1 @ WlE2^T + b2 ----
      v8f d = {};
#pragma unroll
      for (int kb = 0; kb < 2; ++kb) {
        v16h a = *(const v16h*)(hA + (kb << 9) + lane * 16);
        d = wmma32f16(a, loadB(WlE2, 64, c0, kb, lane), d);
      }
#pragma unroll
      for (int r = 0; r < 8; ++r) {
        int idx = (rbase + r) * 64 + ch64;
        r64[idx] = r64[idx] + d[r] + bb2;
      }
    }
    __syncthreads();
    {  // ---- LN2 -> store E tile ----
      float v[4], s1 = 0.f, s2 = 0.f;
#pragma unroll
      for (int c2 = 0; c2 < 4; ++c2) {
        v[c2] = r64[lnedge * 64 + lnc + c2];
        s1 += v[c2]; s2 += v[c2] * v[c2];
      }
#pragma unroll
      for (int o = 1; o < 16; o <<= 1) { s1 += __shfl_xor(s1, o, 16); s2 += __shfl_xor(s2, o, 16); }
      float mu = s1 * (1.f / 64.f);
      float rstd = rsqrtf(s2 * (1.f / 64.f) - mu * mu + 1e-5f);
      float4 ov;
      ov.x = (v[0] - mu) * rstd * g2[0] + o2[0];
      ov.y = (v[1] - mu) * rstd * g2[1] + o2[1];
      ov.z = (v[2] - mu) * rstd * g2[2] + o2[2];
      ov.w = (v[3] - mu) * rstd * g2[3] + o2[3];
      *(float4*)(outE + ((size_t)(i * 768 + j0 + lnedge) * 64 + lnc)) = ov;
    }
  }
  __syncthreads();
  if (tid < 128) wV[i * 128 + tid] = sa[tid] / ss[tid];
}

// ---------------- fused x-side tail ----------------
// grid 48: 16-row tiles. newX + residual + LN + FFN-X (512) + LN -> X.
__global__ void __launch_bounds__(256) k_xout(
    const float* __restrict__ x, const float* __restrict__ wV,
    const _Float16* __restrict__ Wxout, const _Float16* __restrict__ WlX1,
    const _Float16* __restrict__ WlX2,
    const float* __restrict__ bxout, const float* __restrict__ blX1,
    const float* __restrict__ blX2,
    const float* __restrict__ gX1, const float* __restrict__ bX1v,
    const float* __restrict__ gX2, const float* __restrict__ bX2v,
    float* __restrict__ outX) {
  __shared__ __align__(32) _Float16 aA[2048];   // K=128 A layout (wV, then x1)
  __shared__ __align__(32) _Float16 hxA[8192];  // K=512 A layout (relu ff1)
  __shared__ __align__(32) float    r128[2048]; // [16][128] working
  const int m0 = blockIdx.x * 16;
  const int tid = threadIdx.x, lane = tid & 31, wid = tid >> 5;
  const int col = lane & 15, rbase = (lane >> 4) << 3;
  const int n0 = wid * 16, ch = n0 + col;
  for (int idx = tid; idx < 2048; idx += 256) {
    int row = idx >> 7, k = idx & 127;
    aA[((k >> 5) << 9) + aidx(row, k & 31)] = (_Float16)wV[(m0 + row) * 128 + k];
  }
  __syncthreads();
  {  // newX + residual
    v8f d = {};
#pragma unroll
    for (int kb = 0; kb < 4; ++kb) {
      v16h a = *(const v16h*)(aA + (kb << 9) + lane * 16);
      d = wmma32f16(a, loadB(Wxout, 128, n0, kb, lane), d);
    }
    float bb = bxout[ch];
#pragma unroll
    for (int r = 0; r < 8; ++r) {
      int row = rbase + r;
      r128[row * 128 + ch] = x[(m0 + row) * 128 + ch] + d[r] + bb;
    }
  }
  __syncthreads();
  const int sub = tid & 15, lrow = tid >> 4, lc = sub * 8;
  {  // LN1 over 128 (16 threads/row, 8 ch each)
    float v[8], s1 = 0.f, s2 = 0.f;
#pragma unroll
    for (int c2 = 0; c2 < 8; ++c2) { v[c2] = r128[lrow * 128 + lc + c2]; s1 += v[c2]; s2 += v[c2] * v[c2]; }
#pragma unroll
    for (int o = 1; o < 16; o <<= 1) { s1 += __shfl_xor(s1, o, 16); s2 += __shfl_xor(s2, o, 16); }
    float mu = s1 * (1.f / 128.f);
    float rstd = rsqrtf(s2 * (1.f / 128.f) - mu * mu + 1e-5f);
#pragma unroll
    for (int c2 = 0; c2 < 8; ++c2) {
      float xv = (v[c2] - mu) * rstd * gX1[lc + c2] + bX1v[lc + c2];
      int k = lc + c2;
      r128[lrow * 128 + k] = xv;
      aA[((k >> 5) << 9) + aidx(lrow, k & 31)] = (_Float16)xv;
    }
  }
  __syncthreads();
#pragma unroll
  for (int t = 0; t < 4; ++t) {  // ff1: 512 cols, each wave 4 tiles
    int nn0 = t * 128 + wid * 16;
    v8f d = {};
#pragma unroll
    for (int kb = 0; kb < 4; ++kb) {
      v16h a = *(const v16h*)(aA + (kb << 9) + lane * 16);
      d = wmma32f16(a, loadB(WlX1, 128, nn0, kb, lane), d);
    }
    int nch = nn0 + col;
    float bb = blX1[nch];
#pragma unroll
    for (int r = 0; r < 8; ++r) {
      float hv = fmaxf(d[r] + bb, 0.f);
      hxA[((nch >> 5) << 9) + aidx(rbase + r, nch & 31)] = (_Float16)hv;
    }
  }
  __syncthreads();
  {  // ff2 + residual
    v8f d = {};
#pragma unroll
    for (int kb = 0; kb < 16; ++kb) {
      v16h a = *(const v16h*)(hxA + (kb << 9) + lane * 16);
      d = wmma32f16(a, loadB(WlX2, 512, n0, kb, lane), d);
    }
    float bb = blX2[ch];
#pragma unroll
    for (int r = 0; r < 8; ++r) {
      int row = rbase + r;
      r128[row * 128 + ch] = r128[row * 128 + ch] + d[r] + bb;
    }
  }
  __syncthreads();
  {  // LN2 -> X
    float v[8], s1 = 0.f, s2 = 0.f;
#pragma unroll
    for (int c2 = 0; c2 < 8; ++c2) { v[c2] = r128[lrow * 128 + lc + c2]; s1 += v[c2]; s2 += v[c2] * v[c2]; }
#pragma unroll
    for (int o = 1; o < 16; o <<= 1) { s1 += __shfl_xor(s1, o, 16); s2 += __shfl_xor(s2, o, 16); }
    float mu = s1 * (1.f / 128.f);
    float rstd = rsqrtf(s2 * (1.f / 128.f) - mu * mu + 1e-5f);
#pragma unroll
    for (int c2 = 0; c2 < 8; ++c2)
      outX[(m0 + lrow) * 128 + lc + c2] = (v[c2] - mu) * rstd * gX2[lc + c2] + bX2v[lc + c2];
  }
}

// ---------------- host launcher ----------------
extern "C" void kernel_launch(void* const* d_in, const int* in_sizes, int n_in,
                              void* d_out, int out_size, void* d_ws, size_t ws_size,
                              hipStream_t stream) {
  const float* x   = (const float*)d_in[0];
  const float* e   = (const float*)d_in[1];
  const float* Wq  = (const float*)d_in[3];  const float* bq  = (const float*)d_in[4];
  const float* Wk  = (const float*)d_in[5];  const float* bk  = (const float*)d_in[6];
  const float* Wv  = (const float*)d_in[7];  const float* bv  = (const float*)d_in[8];
  const float* Wea = (const float*)d_in[9];  const float* bea = (const float*)d_in[10]; // e_add
  const float* Wem = (const float*)d_in[11]; const float* bem = (const float*)d_in[12]; // e_mul
  const float* Wxo = (const float*)d_in[13]; const float* bxo = (const float*)d_in[14];
  const float* Weo = (const float*)d_in[15]; const float* beo = (const float*)d_in[16];
  const float* Wx1 = (const float*)d_in[17]; const float* bx1 = (const float*)d_in[18];
  const float* Wx2 = (const float*)d_in[19]; const float* bx2 = (const float*)d_in[20];
  const float* We1 = (const float*)d_in[21]; const float* be1 = (const float*)d_in[22];
  const float* We2 = (const float*)d_in[23]; const float* be2 = (const float*)d_in[24];
  const float* gX1 = (const float*)d_in[25]; const float* oX1 = (const float*)d_in[26];
  const float* gX2 = (const float*)d_in[27]; const float* oX2 = (const float*)d_in[28];
  const float* gE1 = (const float*)d_in[29]; const float* oE1 = (const float*)d_in[30];
  const float* gE2 = (const float*)d_in[31]; const float* oE2 = (const float*)d_in[32];

  char* ws = (char*)d_ws;
  size_t off = 0;
  auto take = [&](size_t bytes) -> void* {
    void* p = ws + off;
    off += (bytes + 255) & ~(size_t)255;
    return p;
  };
  _Float16* hWq  = (_Float16*)take(16384 * 2);
  _Float16* hWk  = (_Float16*)take(16384 * 2);
  _Float16* hWv  = (_Float16*)take(16384 * 2);
  _Float16* hWem = (_Float16*)take(8192 * 2);
  _Float16* hWea = (_Float16*)take(8192 * 2);
  _Float16* hWeo = (_Float16*)take(8192 * 2);
  _Float16* hWe1 = (_Float16*)take(4096 * 2);
  _Float16* hWe2 = (_Float16*)take(4096 * 2);
  _Float16* hWxo = (_Float16*)take(16384 * 2);
  _Float16* hWx1 = (_Float16*)take(65536 * 2);
  _Float16* hWx2 = (_Float16*)take(65536 * 2);
  _Float16* hx   = (_Float16*)take(98304 * 2);
  float* Qs = (float*)take(98304 * 4);
  float* Kf = (float*)take(98304 * 4);
  float* Vf = (float*)take(98304 * 4);
  float* wV = (float*)take(98304 * 4);

  auto conv = [&](const float* s, _Float16* d, int n) {
    k_f32_to_f16<<<(n + 255) / 256, 256, 0, stream>>>(s, d, n);
  };
  conv(Wq, hWq, 16384);  conv(Wk, hWk, 16384);  conv(Wv, hWv, 16384);
  conv(Wem, hWem, 8192); conv(Wea, hWea, 8192); conv(Weo, hWeo, 8192);
  conv(We1, hWe1, 4096); conv(We2, hWe2, 4096);
  conv(Wxo, hWxo, 16384); conv(Wx1, hWx1, 65536); conv(Wx2, hWx2, 65536);
  conv(x, hx, 98304);

  k_qkv<<<dim3(48, 3), 256, 0, stream>>>(hx, hWq, hWk, hWv, bq, bk, bv, Qs, Kf, Vf);

  float* outX = (float*)d_out;
  float* outE = outX + 98304;
  k_edge<<<768, 256, 0, stream>>>(e, hWem, hWea, hWeo, hWe1, hWe2,
                                  bem, bea, beo, be1, be2,
                                  gE1, oE1, gE2, oE2,
                                  Qs, Kf, Vf, wV, outE);
  k_xout<<<48, 256, 0, stream>>>(x, wV, hWxo, hWx1, hWx2, bxo, bx1, bx2,
                                 gX1, oX1, gX2, oX2, outX);
}
